// StructureCrossAttention_35029753266947
// MI455X (gfx1250) — matrix operand order repaired and checked by reading, not compiled, not run on hardware
//
#include <hip/hip_runtime.h>
#include <hip/hip_fp16.h>
#include <math.h>
#include <stdint.h>

typedef __attribute__((ext_vector_type(16))) _Float16 v16h;
typedef __attribute__((ext_vector_type(8)))  _Float16 v8h;
typedef __attribute__((ext_vector_type(8)))  float    v8f;
typedef __attribute__((ext_vector_type(4)))  float    v4f;

#define BATCH   16
#define SEQ     256
#define DMODEL  192
#define NHEADS  4
#define DHEAD   48
#define DHPAD   64
#define NRBF    16
#define PFDIM   128
#define ROWS    (BATCH*SEQ)           /* 4096 */
#define QK_SCALE 0.14433756729740643f /* 1/sqrt(48) */
#define RBF_STEP 3.3333333333333335f  /* 50/15 */
#define RBF_INV2W2 0.18f              /* 9/50 */

// ---------------------------------------------------------------------------
// WMMA helpers (CDNA5 wave32, 16x16x32 f16 -> f32)
// ---------------------------------------------------------------------------
__device__ __forceinline__ v8f wmma16(v16h a, v16h b, v8f c) {
  return __builtin_amdgcn_wmma_f32_16x16x32_f16(
      /*neg_a=*/false, a, /*neg_b=*/false, b,
      /*c_mod=*/(short)0, c, /*reuse_a=*/false, /*reuse_b=*/false);
}

// A fragment (16x32, f16): lane holds row (lane&15); 16 contiguous K values
// starting at kbase + (lane>=16 ? 16 : 0).
__device__ __forceinline__ v16h load_a16(const _Float16* rowptr, int kbase) {
  int hi = (threadIdx.x & 16) ? 16 : 0;
  const _Float16* p = rowptr + kbase + hi;
  v8h lo = *(const v8h*)(p);
  v8h up = *(const v8h*)(p + 8);
  v16h a;
#pragma unroll
  for (int i = 0; i < 8; ++i) { a[i] = lo[i]; a[i + 8] = up[i]; }
  return a;
}

__device__ __forceinline__ v16h load_a32(const float* rowptr, int kbase) {
  int hi = (threadIdx.x & 16) ? 16 : 0;
  const float* p = rowptr + kbase + hi;
  v4f x0 = *(const v4f*)(p);
  v4f x1 = *(const v4f*)(p + 4);
  v4f x2 = *(const v4f*)(p + 8);
  v4f x3 = *(const v4f*)(p + 12);
  v16h a;
#pragma unroll
  for (int i = 0; i < 4; ++i) {
    a[i]      = (_Float16)x0[i];
    a[4 + i]  = (_Float16)x1[i];
    a[8 + i]  = (_Float16)x2[i];
    a[12 + i] = (_Float16)x3[i];
  }
  return a;
}

// Non-temporal variant for streamed-once data (pfea): TH=NT, don't pollute L2.
__device__ __forceinline__ v16h load_a32_nt(const float* rowptr, int kbase) {
  int hi = (threadIdx.x & 16) ? 16 : 0;
  const float* p = rowptr + kbase + hi;
  v4f x0 = __builtin_nontemporal_load((const v4f*)(p));
  v4f x1 = __builtin_nontemporal_load((const v4f*)(p + 4));
  v4f x2 = __builtin_nontemporal_load((const v4f*)(p + 8));
  v4f x3 = __builtin_nontemporal_load((const v4f*)(p + 12));
  v16h a;
#pragma unroll
  for (int i = 0; i < 4; ++i) {
    a[i]      = (_Float16)x0[i];
    a[4 + i]  = (_Float16)x1[i];
    a[8 + i]  = (_Float16)x2[i];
    a[12 + i] = (_Float16)x3[i];
  }
  return a;
}

__device__ __forceinline__ v16h load_a(const _Float16* rowptr, int kbase) { return load_a16(rowptr, kbase); }
__device__ __forceinline__ v16h load_a(const float* rowptr, int kbase)    { return load_a32(rowptr, kbase); }

// B fragment (32x16, f16): lane holds column (lane&15); 16 contiguous K values
// starting at kbase + (lane>=16 ? 16 : 0); `p` already includes everything.
template <typename PT>
__device__ __forceinline__ v16h load_b_contig(PT p) {
  v8h lo = *(const v8h*)(p);
  v8h up = *(const v8h*)(p + 8);
  v16h b;
#pragma unroll
  for (int i = 0; i < 8; ++i) { b[i] = lo[i]; b[i + 8] = up[i]; }
  return b;
}

// ---------------------------------------------------------------------------
// Tensor Data Mover staging (gfx1250): 1-row tile of N b16 elements -> LDS.
// D# per CDNA5 ISA. This toolchain exposes the 6-arg
// builtin: (u32x4 g0, i32x8 g1, i32x4 g2, i32x4 g3, i32x8, i32 cpol).
// Falls back to cooperative loads if the builtin is unavailable.
// ---------------------------------------------------------------------------
#if __has_builtin(__builtin_amdgcn_tensor_load_to_lds)
#define HAVE_TDM 1
typedef __attribute__((ext_vector_type(4))) unsigned int u32x4;
typedef __attribute__((ext_vector_type(8))) int i32x8;
typedef __attribute__((ext_vector_type(4))) int i32x4;

__device__ __forceinline__ void tdm_load_b16_row(unsigned lds_byte_off,
                                                 const void* gptr,
                                                 unsigned nelem) {
  unsigned long long ga = (unsigned long long)(uintptr_t)gptr;
  u32x4 g0;
  g0[0] = 1u;                                   // count=1 (user descriptor)
  g0[1] = lds_byte_off;                         // lds_addr (bytes)
  g0[2] = (unsigned)(ga & 0xffffffffu);         // global_addr[31:0]
  g0[3] = (unsigned)((ga >> 32) & 0x01ffffffu)  // global_addr[56:32]
        | (2u << 30);                           // type = 2 ("image")
  i32x8 g1;
  g1[0] = (int)(1u << 16);                      // data_size=1 (2B); wg_mask=0
  g1[1] = (int)((nelem & 0xffffu) << 16);       // tensor_dim0[15:0]
  g1[2] = (int)((nelem >> 16) & 0xffffu)        // tensor_dim0[31:16]
        | (int)(1u << 16);                      // tensor_dim1 = 1
  g1[3] = (int)((nelem & 0xffffu) << 16);       // tile_dim0
  g1[4] = 1;                                    // tile_dim1 = 1
  g1[5] = (int)nelem;                           // tensor_dim0_stride[31:0]
  g1[6] = 0;
  g1[7] = 0;
  __builtin_amdgcn_tensor_load_to_lds(g0, g1, (i32x4)0, (i32x4)0, (i32x8)0, 0);
}
#else
#define HAVE_TDM 0
#endif

// ---------------------------------------------------------------------------
// Weight transpose + convert: W[K][N] (jax [in,out]) -> WT[Npad][K] f16
// ---------------------------------------------------------------------------
__global__ void cvt_wT_kernel(const float* __restrict__ W, _Float16* __restrict__ WT,
                              int K, int N, int Npad) {
  int idx = blockIdx.x * blockDim.x + threadIdx.x;
  int total = Npad * K;
  if (idx >= total) return;
  int n = idx / K, k = idx - n * K;
  WT[idx] = (n < N) ? (_Float16)W[(size_t)k * N + n] : (_Float16)0.0f;
}

// ---------------------------------------------------------------------------
// LayerNorm: wave per row, D=192 (6 elements/lane)
// ---------------------------------------------------------------------------
__global__ __launch_bounds__(256) void ln_kernel(const float* __restrict__ x,
                                                 const float* __restrict__ g,
                                                 const float* __restrict__ bta,
                                                 float* __restrict__ out) {
  int row  = blockIdx.x * 8 + (threadIdx.x >> 5);
  int lane = threadIdx.x & 31;
  const float* xr = x + (size_t)row * DMODEL;
  float v[6];
  float s = 0.f;
#pragma unroll
  for (int i = 0; i < 6; ++i) { v[i] = xr[lane + i * 32]; s += v[i]; }
#pragma unroll
  for (int off = 16; off > 0; off >>= 1) s += __shfl_xor(s, off, 32);
  float mean = s * (1.0f / DMODEL);
  float s2 = 0.f;
#pragma unroll
  for (int i = 0; i < 6; ++i) { float d = v[i] - mean; s2 += d * d; }
#pragma unroll
  for (int off = 16; off > 0; off >>= 1) s2 += __shfl_xor(s2, off, 32);
  float inv = rsqrtf(s2 * (1.0f / DMODEL) + 1e-5f);
  float* orow = out + (size_t)row * DMODEL;
#pragma unroll
  for (int i = 0; i < 6; ++i) {
    int c = lane + i * 32;
    orow[c] = (v[i] - mean) * inv * g[c] + bta[c];
  }
}

// ---------------------------------------------------------------------------
// Generic WMMA GEMM: out[M][N] = A[M][K] @ WT[N][K]^T + bias (+ epilogue)
//   MODE 0: f32 out = acc + bias
//   MODE 1: f32 out = acc + bias + res
//   MODE 2: f16 out = gelu(acc + bias)
//   MODE 3: f32 out = (acc + bias + res) * mask[row]
// ---------------------------------------------------------------------------
__device__ __forceinline__ float gelu_exact(float x) {
  return 0.5f * x * (1.0f + erff(x * 0.70710678118654752f));
}

template <typename AT, int MODE>
__global__ __launch_bounds__(128) void gemm_kernel(
    const AT* __restrict__ A, int lda,
    const _Float16* __restrict__ WT, int K, int N,
    const float* __restrict__ bias,
    float* __restrict__ outF, _Float16* __restrict__ outH,
    const float* __restrict__ res, const unsigned char* __restrict__ maskq) {
  int m0   = blockIdx.x * 16;
  int wave = threadIdx.x >> 5;
  int lane = threadIdx.x & 31;
  int row  = lane & 15;
  int hi8  = (lane & 16) ? 8 : 0;
  int hi16 = (lane & 16) ? 16 : 0;
  const AT* arow = A + (size_t)(m0 + row) * lda;
  int ntiles = N >> 4, ksteps = K >> 5;
  for (int nt = wave; nt < ntiles; nt += 4) {
    v8f acc = {};
    const _Float16* wrow = WT + (size_t)(nt * 16 + row) * K;
    for (int ks = 0; ks < ksteps; ++ks) {
      v16h af = load_a(arow, ks * 32);
      v16h bf = load_b_contig(wrow + ks * 32 + hi16);
      acc = wmma16(af, bf, acc);
    }
    int col = nt * 16 + row;
    float bb = bias[col];
#pragma unroll
    for (int i = 0; i < 8; ++i) {
      int r = m0 + i + hi8;
      float v = acc[i] + bb;
      if (MODE == 0) {
        outF[(size_t)r * N + col] = v;
      } else if (MODE == 1) {
        outF[(size_t)r * N + col] = v + res[(size_t)r * N + col];
      } else if (MODE == 2) {
        outH[(size_t)r * N + col] = (_Float16)gelu_exact(v);
      } else {
        float m = maskq[r] ? 1.0f : 0.0f;
        outF[(size_t)r * N + col] = (v + res[(size_t)r * N + col]) * m;
      }
    }
  }
}

// ---------------------------------------------------------------------------
// Pack Q/K f32 [ROWS][192] -> f16 [ROWS][4][64] head-padded (pad zeros),
// 1/sqrt(dh) folded into Q.
// ---------------------------------------------------------------------------
__global__ void packqk_kernel(const float* __restrict__ src, _Float16* __restrict__ dst,
                              float scale) {
  int idx = blockIdx.x * 256 + threadIdx.x;
  if (idx >= ROWS * NHEADS * DHPAD) return;
  int row = idx >> 8;
  int c   = idx & 255;
  int h = c >> 6, f = c & 63;
  dst[idx] = (f < DHEAD) ? (_Float16)(src[(size_t)row * DMODEL + h * DHEAD + f] * scale)
                         : (_Float16)0.0f;
}

// V f32 [ROWS][192] -> Vt f16 [B][H][48][256] (contiguous in key)
__global__ void packvt_kernel(const float* __restrict__ Vf, _Float16* __restrict__ Vt) {
  int idx = blockIdx.x * 256 + threadIdx.x;
  if (idx >= BATCH * NHEADS * DHEAD * SEQ) return;
  int key  = idx & 255;
  int rest = idx >> 8;
  int f = rest % DHEAD; rest /= DHEAD;
  int h = rest & 3;
  int b = rest >> 2;
  Vt[idx] = (_Float16)Vf[(size_t)(b * SEQ + key) * DMODEL + h * DHEAD + f];
}

// ---------------------------------------------------------------------------
// Fused attention kernel: one block per (b, 16-q tile), 256 threads (8 waves).
// Streams k in chunks of 32 (2 ktiles):
//   * all waves: pfea MLP (WMMA) for each 16q x 16k tile -> per-head bias phs
//   * waves 0..3 (one head each): QK^T (WMMA) + RBF + phs + mask -> logits,
//     online softmax (running m/l in registers; logits & O share the C-layout
//     row<->lane mapping so rescale is register-local), p@V (WMMA, K=32)
// ---------------------------------------------------------------------------
__global__ __launch_bounds__(256) void fused_attn_kernel(
    const float* __restrict__ pfea,
    const _Float16* __restrict__ w1T_g,   // [64][128]
    const _Float16* __restrict__ w2T_g,   // [16][64] padded
    const float* __restrict__ pfea_b1,    // [64]
    const float* __restrict__ dist_w,     // [16][4]
    const _Float16* __restrict__ Qp,      // [B][SEQ][H][64] pre-scaled
    const _Float16* __restrict__ Kp,      // [B][SEQ][H][64]
    const _Float16* __restrict__ Vt,      // [B][H][48][SEQ]
    const float* __restrict__ ca_q, const float* __restrict__ ca_kv,
    const unsigned char* __restrict__ mask_kv,
    float* __restrict__ aout)             // [ROWS][192]
{
  __shared__ _Float16 w1T[64 * 128];
  __shared__ _Float16 w2T[16 * 64];
  __shared__ _Float16 hid[256 * 64];
  __shared__ float    phs[256 * 4];
  __shared__ _Float16 pbuf[NHEADS][16][32];
  __shared__ float    caq_s[16 * 3], cak_s[16 * 3];
  __shared__ float    dw_s[NRBF * NHEADS];
  __shared__ float    b1_s[64];

  int qt = blockIdx.x, b = blockIdx.y;
  int q0 = qt * 16;
  int tid = threadIdx.x, wave = tid >> 5, lane = tid & 31;
  int colx = lane & 15;
  int hi8  = (lane & 16) ? 8 : 0;
  int hi16 = (lane & 16) ? 16 : 0;
  int m0 = wave * 2;     // this wave's pair M-tiles (2 of 16)
  int h  = wave;         // head (waves 0..3 only)

#if HAVE_TDM
  if (wave == 0) {
    tdm_load_b16_row((unsigned)(uintptr_t)(void*)w1T, w1T_g, 64 * 128);
    tdm_load_b16_row((unsigned)(uintptr_t)(void*)w2T, w2T_g, 16 * 64);
    __builtin_amdgcn_s_wait_tensorcnt(0);
  }
#else
  for (int i = tid; i < 64 * 128; i += 256) w1T[i] = w1T_g[i];
  for (int i = tid; i < 16 * 64; i += 256)  w2T[i] = w2T_g[i];
#endif
  if (tid < 48) caq_s[tid] = ca_q[(size_t)(b * SEQ + q0) * 3 + tid];
  if (tid < 64) { dw_s[tid] = dist_w[tid]; b1_s[tid] = pfea_b1[tid]; }

  // Head-wave persistent state
  v8f  Oacc[3];
  v16h aq0, aq1;
  float m8[8], l8[8];
  if (wave < NHEADS) {
#pragma unroll
    for (int nt = 0; nt < 3; ++nt) Oacc[nt] = (v8f){};
#pragma unroll
    for (int i = 0; i < 8; ++i) { m8[i] = -INFINITY; l8[i] = 0.f; }
    const _Float16* qrow = Qp + ((size_t)((b * SEQ + q0 + colx) * NHEADS + h)) * DHPAD;
    aq0 = load_a16(qrow, 0);
    aq1 = load_a16(qrow, 32);
  }

  // One 16q x 16k tile: pfea MLP bias + QK^T logits into Sout (head waves).
  auto tile_phase = [&](int kt, v8f& Sout) {
    int k0 = kt * 16;
    __syncthreads();   // protect phs/cak from previous tile's readers

    if (tid < 48) cak_s[tid] = ca_kv[(size_t)(b * SEQ + k0) * 3 + tid];

    // ---- pfea GEMM1: [256 pairs x 128] @ [128 x 64] (all waves)
    v8f acc[2][4];
#pragma unroll
    for (int j = 0; j < 2; ++j)
#pragma unroll
      for (int n = 0; n < 4; ++n) acc[j][n] = (v8f){};
    const float* arow0 = pfea + ((size_t)((b * SEQ + q0 + m0    ) * SEQ + k0 + colx)) * PFDIM;
    const float* arow1 = pfea + ((size_t)((b * SEQ + q0 + m0 + 1) * SEQ + k0 + colx)) * PFDIM;
    for (int ks = 0; ks < 4; ++ks) {
      v16h a0 = load_a32_nt(arow0, ks * 32);
      v16h a1 = load_a32_nt(arow1, ks * 32);
#pragma unroll
      for (int n = 0; n < 4; ++n) {
        v16h bf = load_b_contig(&w1T[(n * 16 + colx) * 128 + ks * 32 + hi16]);
        acc[0][n] = wmma16(a0, bf, acc[0][n]);
        acc[1][n] = wmma16(a1, bf, acc[1][n]);
      }
    }
#pragma unroll
    for (int j = 0; j < 2; ++j)
#pragma unroll
      for (int n = 0; n < 4; ++n) {
        int feat = n * 16 + colx;
        float bb = b1_s[feat];
#pragma unroll
        for (int i = 0; i < 8; ++i) {
          int pr = (m0 + j) * 16 + i + hi8;   // pair index (q'*16 + k')
          hid[pr * 64 + feat] = (_Float16)fmaxf(acc[j][n][i] + bb, 0.0f);
        }
      }
    // hid rows for this wave's M-tiles were written by this wave; same-wave
    // LDS ops are in order -> no barrier before GEMM2.

    // ---- pfea GEMM2: [256 x 64] @ [64 x 16(pad, 4 heads)]
    v8f acc2[2];
    acc2[0] = (v8f){}; acc2[1] = (v8f){};
    for (int ks = 0; ks < 2; ++ks) {
      v16h a0 = load_a16(&hid[((m0    ) * 16 + colx) * 64], ks * 32);
      v16h a1 = load_a16(&hid[((m0 + 1) * 16 + colx) * 64], ks * 32);
      v16h bf = load_b_contig(&w2T[colx * 64 + ks * 32 + hi16]);
      acc2[0] = wmma16(a0, bf, acc2[0]);
      acc2[1] = wmma16(a1, bf, acc2[1]);
    }
#pragma unroll
    for (int j = 0; j < 2; ++j) {
      if (colx < NHEADS) {
#pragma unroll
        for (int i = 0; i < 8; ++i) {
          int pr = (m0 + j) * 16 + i + hi8;
          phs[pr * 4 + colx] = acc2[j][i];
        }
      }
    }
    __syncthreads();   // phs + cak ready for head waves

    // ---- head waves: logits for this 16x16 tile
    if (wave < NHEADS) {
      v8f accS = (v8f){};
      const _Float16* krow = Kp + ((size_t)((b * SEQ + k0 + colx) * NHEADS + h)) * DHPAD;
      accS = wmma16(aq0, load_b_contig(krow + hi16), accS);
      accS = wmma16(aq1, load_b_contig(krow + 32 + hi16), accS);
#pragma unroll
      for (int i = 0; i < 8; ++i) {
        int qi = i + hi8;
        int ki = colx;
        float dx = caq_s[qi * 3 + 0] - cak_s[ki * 3 + 0];
        float dy = caq_s[qi * 3 + 1] - cak_s[ki * 3 + 1];
        float dz = caq_s[qi * 3 + 2] - cak_s[ki * 3 + 2];
        float dist = sqrtf(dx * dx + dy * dy + dz * dz);
        float bias = 0.f;
#pragma unroll
        for (int r = 0; r < NRBF; ++r) {
          float d = dist - (float)r * RBF_STEP;
          bias += __expf(-d * d * RBF_INV2W2) * dw_s[r * NHEADS + h];
        }
        float val = accS[i] + bias + phs[(qi * 16 + ki) * 4 + h];
        if (!mask_kv[b * SEQ + k0 + ki]) val = -INFINITY;
        Sout[i] = val;
      }
    }
  };

  for (int chunk = 0; chunk < 8; ++chunk) {
    v8f S0, S1;
    tile_phase(chunk * 2 + 0, S0);
    tile_phase(chunk * 2 + 1, S1);

    // ---- online softmax over the 32-column chunk + p@V (head waves)
    if (wave < NHEADS) {
      float p0[8], p1[8], resc[8];
#pragma unroll
      for (int i = 0; i < 8; ++i) {
        float mrow = fmaxf(S0[i], S1[i]);
#pragma unroll
        for (int off = 8; off > 0; off >>= 1) mrow = fmaxf(mrow, __shfl_xor(mrow, off, 32));
        float mnew = fmaxf(m8[i], mrow);
        bool  live = mnew > -INFINITY;
        float sc = live ? __expf(m8[i] - mnew) : 1.0f;
        float a0 = live ? __expf(S0[i] - mnew) : 0.0f;
        float a1 = live ? __expf(S1[i] - mnew) : 0.0f;
        float rs = a0 + a1;
#pragma unroll
        for (int off = 8; off > 0; off >>= 1) rs += __shfl_xor(rs, off, 32);
        l8[i] = l8[i] * sc + rs;
        m8[i] = mnew;
        p0[i] = a0; p1[i] = a1; resc[i] = sc;
      }
      // stage p (per-wave private LDS slice; same-wave DS ops are in order)
#pragma unroll
      for (int i = 0; i < 8; ++i) {
        int r = i + hi8;
        pbuf[h][r][colx]      = (_Float16)p0[i];
        pbuf[h][r][16 + colx] = (_Float16)p1[i];
      }
#pragma unroll
      for (int nt = 0; nt < 3; ++nt)
#pragma unroll
        for (int i = 0; i < 8; ++i) Oacc[nt][i] *= resc[i];
      v16h ap = load_a16(&pbuf[h][colx][0], 0);
#pragma unroll
      for (int nt = 0; nt < 3; ++nt) {
        const _Float16* vb = Vt + ((size_t)(b * NHEADS + h) * DHEAD + nt * 16 + colx) * SEQ
                           + chunk * 32 + hi16;
        Oacc[nt] = wmma16(ap, load_b_contig(vb), Oacc[nt]);
      }
    }
  } // chunk

  // ---- finalize: O / l -> aout
  if (wave < NHEADS) {
    float inv[8];
#pragma unroll
    for (int i = 0; i < 8; ++i) inv[i] = (l8[i] > 0.f) ? 1.0f / l8[i] : 0.f;
#pragma unroll
    for (int nt = 0; nt < 3; ++nt)
#pragma unroll
      for (int i = 0; i < 8; ++i) {
        int r = b * SEQ + q0 + i + hi8;
        aout[(size_t)r * DMODEL + h * DHEAD + nt * 16 + colx] = Oacc[nt][i] * inv[i];
      }
  }
}

// ---------------------------------------------------------------------------
// Launch
// ---------------------------------------------------------------------------
extern "C" void kernel_launch(void* const* d_in, const int* in_sizes, int n_in,
                              void* d_out, int out_size, void* d_ws, size_t ws_size,
                              hipStream_t stream) {
  const float* q_feats  = (const float*)d_in[0];
  const float* kv_feats = (const float*)d_in[1];
  const float* ca_q     = (const float*)d_in[2];
  const float* ca_kv    = (const float*)d_in[3];
  const float* pfea     = (const float*)d_in[4];
  const unsigned char* mask_q  = (const unsigned char*)d_in[5];
  const unsigned char* mask_kv = (const unsigned char*)d_in[6];
  const float* Wq_w = (const float*)d_in[7];   const float* Wq_b = (const float*)d_in[8];
  const float* Wk_w = (const float*)d_in[9];   const float* Wk_b = (const float*)d_in[10];
  const float* Wv_w = (const float*)d_in[11];  const float* Wv_b = (const float*)d_in[12];
  const float* Wo_w = (const float*)d_in[13];  const float* Wo_b = (const float*)d_in[14];
  const float* lnq_g  = (const float*)d_in[15]; const float* lnq_b  = (const float*)d_in[16];
  const float* lnkv_g = (const float*)d_in[17]; const float* lnkv_b = (const float*)d_in[18];
  const float* dist_w = (const float*)d_in[19];
  const float* pfea_w1 = (const float*)d_in[20]; const float* pfea_b1 = (const float*)d_in[21];
  const float* pfea_w2 = (const float*)d_in[22];
  const float* ffnln_g = (const float*)d_in[23]; const float* ffnln_b = (const float*)d_in[24];
  const float* ffn_w1  = (const float*)d_in[25]; const float* ffn_b1  = (const float*)d_in[26];
  const float* ffn_w2  = (const float*)d_in[27]; const float* ffn_b2  = (const float*)d_in[28];

  char* wp = (char*)d_ws;
  auto alloc = [&](size_t bytes) -> void* {
    void* r = (void*)wp;
    wp += (bytes + 255) & ~(size_t)255;
    return r;
  };

  float* qn       = (float*)alloc((size_t)ROWS * DMODEL * 4);
  float* kvn      = (float*)alloc((size_t)ROWS * DMODEL * 4);
  float* Qf       = (float*)alloc((size_t)ROWS * DMODEL * 4);
  float* Kf       = (float*)alloc((size_t)ROWS * DMODEL * 4);
  float* Vf       = (float*)alloc((size_t)ROWS * DMODEL * 4);
  float* attended = (float*)alloc((size_t)ROWS * DMODEL * 4);
  float* ffn_in   = (float*)alloc((size_t)ROWS * DMODEL * 4);
  float* aout     = (float*)alloc((size_t)ROWS * DMODEL * 4);
  _Float16* Qp    = (_Float16*)alloc((size_t)ROWS * NHEADS * DHPAD * 2);
  _Float16* Kp    = (_Float16*)alloc((size_t)ROWS * NHEADS * DHPAD * 2);
  _Float16* Vt    = (_Float16*)alloc((size_t)BATCH * NHEADS * DHEAD * SEQ * 2);
  _Float16* ffn_h = (_Float16*)alloc((size_t)ROWS * 2 * DMODEL * 2);
  _Float16* WqT = (_Float16*)alloc((size_t)DMODEL * DMODEL * 2);
  _Float16* WkT = (_Float16*)alloc((size_t)DMODEL * DMODEL * 2);
  _Float16* WvT = (_Float16*)alloc((size_t)DMODEL * DMODEL * 2);
  _Float16* WoT = (_Float16*)alloc((size_t)DMODEL * DMODEL * 2);
  _Float16* w1T = (_Float16*)alloc((size_t)64 * 128 * 2);
  _Float16* w2T = (_Float16*)alloc((size_t)16 * 64 * 2);
  _Float16* f1T = (_Float16*)alloc((size_t)(2 * DMODEL) * DMODEL * 2);
  _Float16* f2T = (_Float16*)alloc((size_t)DMODEL * (2 * DMODEL) * 2);
  (void)ws_size; (void)in_sizes; (void)n_in; (void)out_size;

  auto cvt = [&](const float* W, _Float16* WT, int K, int N, int Npad) {
    int total = Npad * K;
    cvt_wT_kernel<<<(total + 255) / 256, 256, 0, stream>>>(W, WT, K, N, Npad);
  };
  cvt(Wq_w, WqT, DMODEL, DMODEL, DMODEL);
  cvt(Wk_w, WkT, DMODEL, DMODEL, DMODEL);
  cvt(Wv_w, WvT, DMODEL, DMODEL, DMODEL);
  cvt(Wo_w, WoT, DMODEL, DMODEL, DMODEL);
  cvt(pfea_w1, w1T, 128, 64, 64);
  cvt(pfea_w2, w2T, 64, 4, 16);
  cvt(ffn_w1, f1T, DMODEL, 2 * DMODEL, 2 * DMODEL);
  cvt(ffn_w2, f2T, 2 * DMODEL, DMODEL, DMODEL);

  // LayerNorms
  ln_kernel<<<ROWS / 8, 256, 0, stream>>>(q_feats, lnq_g, lnq_b, qn);
  ln_kernel<<<ROWS / 8, 256, 0, stream>>>(kv_feats, lnkv_g, lnkv_b, kvn);

  // Q/K/V projections (f32 activations, WMMA, f32 outputs)
  gemm_kernel<float, 0><<<ROWS / 16, 128, 0, stream>>>(qn,  DMODEL, WqT, DMODEL, DMODEL, Wq_b, Qf, nullptr, nullptr, nullptr);
  gemm_kernel<float, 0><<<ROWS / 16, 128, 0, stream>>>(kvn, DMODEL, WkT, DMODEL, DMODEL, Wk_b, Kf, nullptr, nullptr, nullptr);
  gemm_kernel<float, 0><<<ROWS / 16, 128, 0, stream>>>(kvn, DMODEL, WvT, DMODEL, DMODEL, Wv_b, Vf, nullptr, nullptr, nullptr);

  // Pack Q (pre-scaled), K (padded heads), V (transposed)
  packqk_kernel<<<(ROWS * NHEADS * DHPAD + 255) / 256, 256, 0, stream>>>(Qf, Qp, QK_SCALE);
  packqk_kernel<<<(ROWS * NHEADS * DHPAD + 255) / 256, 256, 0, stream>>>(Kf, Kp, 1.0f);
  packvt_kernel<<<(BATCH * NHEADS * DHEAD * SEQ + 255) / 256, 256, 0, stream>>>(Vf, Vt);

  // Fused: pfea-MLP bias + QK^T + RBF + mask + online softmax + p@V
  fused_attn_kernel<<<dim3(SEQ / 16, BATCH), 256, 0, stream>>>(
      pfea, w1T, w2T, pfea_b1, dist_w, Qp, Kp, Vt, ca_q, ca_kv, mask_kv, aout);

  // attended = qn + aout @ Wo + b
  gemm_kernel<float, 1><<<ROWS / 16, 128, 0, stream>>>(aout, DMODEL, WoT, DMODEL, DMODEL, Wo_b, attended, nullptr, qn, nullptr);

  // FFN
  ln_kernel<<<ROWS / 8, 256, 0, stream>>>(attended, ffnln_g, ffnln_b, ffn_in);
  gemm_kernel<float, 2><<<ROWS / 16, 128, 0, stream>>>(ffn_in, DMODEL, f1T, DMODEL, 2 * DMODEL, ffn_b1, nullptr, ffn_h, nullptr, nullptr);
  gemm_kernel<_Float16, 3><<<ROWS / 16, 128, 0, stream>>>(ffn_h, 2 * DMODEL, f2T, 2 * DMODEL, DMODEL, ffn_b2, (float*)d_out, nullptr, attended, mask_q);
}
